// GraphAttentionLayer_55052890800928
// MI455X (gfx1250) — matrix operand order repaired and checked by reading, not compiled, not run on hardware
//
#include <hip/hip_runtime.h>
#include <hip/hip_bf16.h>

typedef __attribute__((ext_vector_type(16))) _Float16 v16h;
typedef __attribute__((ext_vector_type(8)))  float    v8f;

#define N_NODES 10000
#define F_IN    256
#define F_OUT   64
#define NP      10016        /* N padded to multiple of 32 */
#define NEG_BIG (-9.0e15f)
#define ALPHA_L 0.2f

static __device__ __forceinline__ v8f wmma16(v16h a, v16h b, v8f c) {
  // D = A(16x32 f16) * B(32x16 f16) + C(16x16 f32)
  return __builtin_amdgcn_wmma_f32_16x16x32_f16(false, a, false, b, (short)0, c,
                                                false, false);
}

// ---------------------------------------------------------------------------
// Kernel 0: WT[n][k] = (f16) W[k][n]   (64 x 256 transposed f16 copy of W)
// ---------------------------------------------------------------------------
__global__ void __launch_bounds__(256)
k_wt(const float* __restrict__ W, _Float16* __restrict__ WT) {
  const int t = blockIdx.x * blockDim.x + threadIdx.x;   // 0 .. 64*256-1
  if (t >= F_OUT * F_IN) return;
  const int n = t >> 8;          // output column  (0..63)
  const int k = t & 255;         // input feature  (0..255)
  WT[(size_t)n * F_IN + k] = (_Float16)W[(size_t)k * F_OUT + n];
}

// ---------------------------------------------------------------------------
// Kernel 1: Wh = h @ W   (one wave per 16-row tile, WMMA f16)
// Stores Wh f32 row-major [N][64] and WhT f16 column-major [64][NP].
// ---------------------------------------------------------------------------
__global__ void __launch_bounds__(32)
k_wh(const float* __restrict__ h, const _Float16* __restrict__ WT,
     float* __restrict__ Wh, _Float16* __restrict__ WhT) {
  const int i0   = blockIdx.x * 16;
  const int lane = threadIdx.x & 31;
  const int lo   = lane & 15;
  const int hi   = lane >> 4;
  const int row  = i0 + lo;

  v8f acc[4] = {};

  for (int k0 = 0; k0 < F_IN; k0 += 32) {
    // A operand: h tile 16x32, lane layout K(e) = (e/8)*16 + hi*8 + (e%8)
    v16h av;
    const float* hp = h + (size_t)row * F_IN + k0 + hi * 8;
#pragma unroll
    for (int g = 0; g < 2; ++g) {
      float4 x0 = *(const float4*)(hp + g * 16);
      float4 x1 = *(const float4*)(hp + g * 16 + 4);
      av[g * 8 + 0] = (_Float16)x0.x; av[g * 8 + 1] = (_Float16)x0.y;
      av[g * 8 + 2] = (_Float16)x0.z; av[g * 8 + 3] = (_Float16)x0.w;
      av[g * 8 + 4] = (_Float16)x1.x; av[g * 8 + 5] = (_Float16)x1.y;
      av[g * 8 + 6] = (_Float16)x1.z; av[g * 8 + 7] = (_Float16)x1.w;
    }
    // B operand: lane layout K(e) = (e/8)*16 + hi*8 + (e%8); from transposed WT
#pragma unroll
    for (int nt = 0; nt < 4; ++nt) {
      union { uint4 u[2]; v16h v; } bb;
      const _Float16* tp = WT + (size_t)(nt * 16 + lo) * F_IN + k0 + hi * 8;
      bb.u[0] = *(const uint4*)tp;
      bb.u[1] = *(const uint4*)(tp + 16);
      acc[nt] = wmma16(av, bb.v, acc[nt]);
    }
  }

  // D layout: VGPR r, lane<16 -> row r, lane>=16 -> row r+8; col = lane&15
#pragma unroll
  for (int nt = 0; nt < 4; ++nt) {
    const int col = nt * 16 + lo;
    union { _Float16 hh[8]; uint4 u; } pk;
#pragma unroll
    for (int r = 0; r < 8; ++r) {
      float v = acc[nt][r];
      int rr = i0 + (hi ? r + 8 : r);
      Wh[(size_t)rr * F_OUT + col] = v;
      pk.hh[r] = (_Float16)v;
    }
    *(uint4*)(WhT + (size_t)col * NP + i0 + hi * 8) = pk.u;
  }
}

// ---------------------------------------------------------------------------
// Kernel 2: f1[j] = Wh[j,:] . a[:64],  f2[j] = Wh[j,:] . a[64:].
// Also zeroes WhT padding columns (j in [N, NP)).
// ---------------------------------------------------------------------------
__global__ void __launch_bounds__(128)
k_f12(const float* __restrict__ Wh, const float* __restrict__ a,
      float* __restrict__ f1, float* __restrict__ f2,
      _Float16* __restrict__ WhT) {
  const int j = blockIdx.x * blockDim.x + threadIdx.x;
  if (j >= NP) return;
  if (j < N_NODES) {
    float s1 = 0.f, s2 = 0.f;
    const float* wp = Wh + (size_t)j * F_OUT;
#pragma unroll 8
    for (int n = 0; n < F_OUT; ++n) {
      float v = wp[n];
      s1 += v * a[n];
      s2 += v * a[F_OUT + n];
    }
    f1[j] = s1;
    f2[j] = s2;
  } else {
    f1[j] = 0.f;
    f2[j] = 0.f;
#pragma unroll
    for (int n = 0; n < F_OUT; ++n) WhT[(size_t)n * NP + j] = (_Float16)0.f;
  }
}

// ---------------------------------------------------------------------------
// Kernel 3: fused masked softmax + (attention @ Wh) + ELU, flash-style online
// softmax. One wave handles 16 output rows; j swept in chunks of 32.
// ---------------------------------------------------------------------------
__global__ void __launch_bounds__(32)
k_attn(const int* __restrict__ adj, const float* __restrict__ f1,
       const float* __restrict__ f2, const _Float16* __restrict__ WhT,
       float* __restrict__ out) {
  const int i0   = blockIdx.x * 16;
  const int lane = threadIdx.x & 31;
  const int lo   = lane & 15;
  const int hi   = lane >> 4;
  const int row  = i0 + lo;

  const int* adjRow = adj + (size_t)row * N_NODES;
  const float f1r = f1[row];

  float m = NEG_BIG;   // running row max (row = lane&15; lanes L and L+16 agree)
  float s = 0.f;       // running partial sum of exp (per lane; pair-summed later)
  v8f acc[4] = {};

  for (int j0 = 0; j0 < NP; j0 += 32) {
    float ev[16];
    bool vr0, vr1;
    // ---- compute e values in WMMA-A lane order: K(e) = (e/8)*16 + hi*8 + e%8
#pragma unroll
    for (int g = 0; g < 2; ++g) {
      const int jb = j0 + g * 16 + hi * 8;       // 8-aligned run; N%8==0
      const bool vr = jb < N_NODES;
      if (g == 0) vr0 = vr; else vr1 = vr;
      int4 a0, a1; float4 q0, q1;
      if (vr) {
        a0 = *(const int4*)(adjRow + jb);
        a1 = *(const int4*)(adjRow + jb + 4);
        q0 = *(const float4*)(f2 + jb);
        q1 = *(const float4*)(f2 + jb + 4);
      } else {
        a0 = make_int4(0, 0, 0, 0); a1 = a0;
        q0 = make_float4(0.f, 0.f, 0.f, 0.f); q1 = q0;
      }
      const int   ad[8] = {a0.x, a0.y, a0.z, a0.w, a1.x, a1.y, a1.z, a1.w};
      const float fv[8] = {q0.x, q0.y, q0.z, q0.w, q1.x, q1.y, q1.z, q1.w};
#pragma unroll
      for (int w = 0; w < 8; ++w) {
        float e = f1r + fv[w];
        e = e > 0.f ? e : ALPHA_L * e;           // LeakyReLU
        ev[g * 8 + w] = (ad[w] != 0) ? e : NEG_BIG;
      }
    }
    __builtin_prefetch(adjRow + j0 + 32, 0, 0);  // global_prefetch_b8

    // ---- chunk max over the lane pair (L, L^16) covering the same row
    float lmax = ev[0];
#pragma unroll
    for (int q = 1; q < 16; ++q) lmax = fmaxf(lmax, ev[q]);
    lmax = fmaxf(lmax, __shfl_xor(lmax, 16, 32));
    const float nm = fmaxf(m, lmax);

    // ---- rescale only when some row's max actually grew (rare after warmup;
    //      when nm==m for every lane the factor is exactly 1 -> exact skip)
    if (__any(nm > m)) {
      const float sc = __expf(m - nm);
      s *= sc;
#pragma unroll
      for (int t = 0; t < 8; ++t) {
        const float fac = __shfl(sc, hi ? t + 8 : t, 32);
#pragma unroll
        for (int nt = 0; nt < 4; ++nt) acc[nt][t] *= fac;
      }
    }

    // ---- probabilities -> f16 A operand
    v16h av;
    float ls = 0.f;
#pragma unroll
    for (int q = 0; q < 16; ++q) {
      const bool vr = (q < 8) ? vr0 : vr1;
      float p = vr ? __expf(ev[q] - nm) : 0.f;   // padded j contribute 0
      ls += p;
      av[q] = (_Float16)p;
    }
    s += ls;
    m = nm;

    // ---- B operands from transposed f16 Wh: K(e) = (e/8)*16 + hi*8 + e%8
#pragma unroll
    for (int nt = 0; nt < 4; ++nt) {
      union { uint4 u[2]; v16h v; } bb;
      const _Float16* tp = WhT + (size_t)(nt * 16 + lo) * NP + j0 + hi * 8;
      bb.u[0] = *(const uint4*)tp;
      bb.u[1] = *(const uint4*)(tp + 16);
      acc[nt] = wmma16(av, bb.v, acc[nt]);
    }
  }

  // ---- normalize, ELU, store
  const float stot = s + __shfl_xor(s, 16, 32);
#pragma unroll
  for (int t = 0; t < 8; ++t) {
    const int rr = hi ? t + 8 : t;               // row index 0..15 within tile
    const float sd  = __shfl(stot, rr, 32);
    const float inv = sd > 0.f ? 1.f / sd : 0.f;
#pragma unroll
    for (int nt = 0; nt < 4; ++nt) {
      float o = acc[nt][t] * inv;
      o = o > 0.f ? o : expm1f(o);               // ELU (alpha = 1)
      out[(size_t)(i0 + rr) * F_OUT + nt * 16 + lo] = o;
    }
  }
}

// ---------------------------------------------------------------------------
extern "C" void kernel_launch(void* const* d_in, const int* in_sizes, int n_in,
                              void* d_out, int out_size, void* d_ws, size_t ws_size,
                              hipStream_t stream) {
  const float* h   = (const float*)d_in[0];   // [N, 256]
  const int*   adj = (const int*)  d_in[1];   // [N, N]
  const float* W   = (const float*)d_in[2];   // [256, 64]
  const float* a   = (const float*)d_in[3];   // [128, 1]
  float* out = (float*)d_out;                 // [N, 64]

  char* ws = (char*)d_ws;
  const size_t off_WhT = 0;                                     // 64*NP*2
  const size_t off_Wh  = off_WhT + (size_t)F_OUT * NP * 2;      // 1,282,048
  const size_t off_f1  = off_Wh + (size_t)N_NODES * F_OUT * 4;  // +2,560,000
  const size_t off_f2  = off_f1 + (size_t)NP * 4;
  const size_t off_WT  = off_f2 + (size_t)NP * 4;               // 64*256*2
  _Float16* WhT = (_Float16*)(ws + off_WhT);
  float*    Wh  = (float*)   (ws + off_Wh);
  float*    f1  = (float*)   (ws + off_f1);
  float*    f2  = (float*)   (ws + off_f2);
  _Float16* WT  = (_Float16*)(ws + off_WT);

  (void)in_sizes; (void)n_in; (void)out_size; (void)ws_size;

  k_wt  <<<(F_OUT * F_IN + 255) / 256, 256, 0, stream>>>(W, WT);
  k_wh  <<<N_NODES / 16, 32, 0, stream>>>(h, WT, Wh, WhT);
  k_f12 <<<(NP + 127) / 128, 128, 0, stream>>>(Wh, a, f1, f2, WhT);
  k_attn<<<N_NODES / 16, 32, 0, stream>>>(adj, f1, f2, WhT, out);
}
